// CrossSparseAttention_70231305224555
// MI455X (gfx1250) — matrix operand order repaired and checked by reading, not compiled, not run on hardware
//
#include <hip/hip_runtime.h>

// Problem constants (fixed by the reference setup_inputs()).
#define N_IMG 64
#define R_DIM 36
#define D_DIM 512
#define N_CAP 64
#define W_DIM 50
#define KNN   5
#define INV_T 10.0f          // 1 / TEMPERATURE
#define EPS_V (-1e-8f)
#define MASK_V (-1.0f)

#define S_PITCH 52           // padded LDS pitch for the 64x48 score tile
#define G_PITCH 37           // padded LDS pitch for the 36x36 Gram tile

typedef __attribute__((ext_vector_type(16))) __bf16 bf16x16;
typedef __attribute__((ext_vector_type(8)))  __bf16 bf16x8;
typedef __attribute__((ext_vector_type(8)))  float  f32x8;

static __device__ __forceinline__ int imin(int a, int b) { return a < b ? a : b; }

// A-matrix 16x32 bf16 fragment (ISA 7.12.2):
//  lane<16 : elems 0..7 = K[k0..k0+7],   elems 8..15 = K[k0+16..k0+23]
//  lane>=16: elems 0..7 = K[k0+8..k0+15],elems 8..15 = K[k0+24..k0+31]
static __device__ __forceinline__ bf16x16 load_a_frag(const __bf16* __restrict__ row,
                                                      int k0, int lhi) {
  bf16x8 lo = *(const bf16x8*)(row + k0 + 8 * lhi);
  bf16x8 hi = *(const bf16x8*)(row + k0 + 16 + 8 * lhi);
  bf16x16 v;
#pragma unroll
  for (int j = 0; j < 8; ++j) { v[j] = lo[j]; v[8 + j] = hi[j]; }
  return v;
}

// B-matrix 32x16 bf16 fragment (same K order as A):
//  lane<16 : elems 0..7 = K[k0..k0+7], 8..15 = K[k0+16..k0+23]; lane>=16 : K[k0+8..k0+15], K[k0+24..k0+31]
static __device__ __forceinline__ bf16x16 load_b_frag(const __bf16* __restrict__ row,
                                                      int k0, int lhi) {
  bf16x8 lo = *(const bf16x8*)(row + k0 + 8 * lhi);
  bf16x8 hi = *(const bf16x8*)(row + k0 + 16 + 8 * lhi);
  bf16x16 v;
#pragma unroll
  for (int j = 0; j < 8; ++j) { v[j] = lo[j]; v[8 + j] = hi[j]; }
  return v;
}

// ---------------------------------------------------------------------------
// Kernel 1: fp32 -> bf16 conversion of caps and imgs into workspace.
// ---------------------------------------------------------------------------
__global__ __launch_bounds__(256) void cvt_kernel(const float* __restrict__ caps,
                                                  const float* __restrict__ imgs,
                                                  unsigned short* __restrict__ caps_bf,
                                                  unsigned short* __restrict__ imgs_bf) {
  const int idx = blockIdx.x * blockDim.x + threadIdx.x;
  const int nc = N_CAP * W_DIM * D_DIM;   // 1,638,400
  const int ni = N_IMG * R_DIM * D_DIM;   // 1,179,648
  if (idx < nc) ((__bf16*)caps_bf)[idx] = (__bf16)caps[idx];
  if (idx < ni) ((__bf16*)imgs_bf)[idx] = (__bf16)imgs[idx];
}

// ---------------------------------------------------------------------------
// Kernel 2: per-image Gram matrix G[i] = imgs[i] * imgs[i]^T  (36x36, f32)
// via v_wmma_f32_16x16x32_bf16.  One block per image, waves 0..2 own a row
// tile of 16, each computes 3 column tiles over K=512.
// ---------------------------------------------------------------------------
__global__ __launch_bounds__(128) void gram_kernel(const unsigned short* __restrict__ imgs_bf_u,
                                                   float* __restrict__ G) {
  const int i    = blockIdx.x;
  const int wave = threadIdx.x >> 5;
  const int lane = threadIdx.x & 31;
  if (wave >= 3) return;
  const int lane_lo = lane & 15;
  const int lane_hi = lane >> 4;

  const __bf16* img = (const __bf16*)imgs_bf_u + (size_t)i * R_DIM * D_DIM;
  const int mrow = wave * 16 + lane_lo;
  const __bf16* arow = img + (size_t)imin(mrow, R_DIM - 1) * D_DIM;

  f32x8 acc[3] = {};
  for (int k0 = 0; k0 < D_DIM; k0 += 32) {
    bf16x16 a = load_a_frag(arow, k0, lane_hi);
#pragma unroll
    for (int rt = 0; rt < 3; ++rt) {
      const int nrow = rt * 16 + lane_lo;
      const __bf16* brow = img + (size_t)imin(nrow, R_DIM - 1) * D_DIM;
      bf16x16 b = load_b_frag(brow, k0, lane_hi);
      acc[rt] = __builtin_amdgcn_wmma_f32_16x16x32_bf16(false, a, false, b,
                                                        (short)0, acc[rt], false, false);
    }
  }

  // C/D layout: lane<16 -> (M=j,   N=lane); lane>=16 -> (M=j+8, N=lane-16)
#pragma unroll
  for (int rt = 0; rt < 3; ++rt) {
#pragma unroll
    for (int j = 0; j < 8; ++j) {
      const int m = wave * 16 + j + 8 * lane_hi;
      const int n = rt * 16 + lane_lo;
      if (m < R_DIM && n < R_DIM)
        G[(size_t)i * R_DIM * R_DIM + m * R_DIM + n] = acc[rt][j];
    }
  }
}

// ---------------------------------------------------------------------------
// Kernel 3: one block per (cap c, img i) pair.
//   Phase A (4 waves): S = caps[c] (50x512) x imgs[i]^T (512x36) via WMMA,
//                      64x48 padded tile spilled to LDS.
//   Phase B (50 threads): per caption-word row: mask, top-5 threshold,
//                      masked softmax; num = sum p*S_raw ; sq = p^T G p.
//   out[i,c,w] = num / (sqrt(sq) + EPS), MASK_VAL where w >= cap_len.
// ---------------------------------------------------------------------------
__global__ __launch_bounds__(128) void attn_kernel(const unsigned short* __restrict__ caps_bf_u,
                                                   const unsigned short* __restrict__ imgs_bf_u,
                                                   const float* __restrict__ Gmat,
                                                   const int* __restrict__ img_lens,
                                                   const int* __restrict__ cap_lens,
                                                   float* __restrict__ out) {
  __shared__ float S[64 * S_PITCH];        // raw dot products (then reused for p-weights)
  __shared__ float Gs[R_DIM * G_PITCH];    // Gram of imgs[i]

  const int c = blockIdx.x;
  const int i = blockIdx.y;
  const int wave    = threadIdx.x >> 5;
  const int lane    = threadIdx.x & 31;
  const int lane_lo = lane & 15;
  const int lane_hi = lane >> 4;

  // Stage G[i] into LDS (independent of the WMMA phase).
  for (int x = threadIdx.x; x < R_DIM * R_DIM; x += blockDim.x)
    Gs[(x / R_DIM) * G_PITCH + (x % R_DIM)] = Gmat[(size_t)i * R_DIM * R_DIM + x];

  const __bf16* cap = (const __bf16*)caps_bf_u + (size_t)c * W_DIM * D_DIM;
  const __bf16* img = (const __bf16*)imgs_bf_u + (size_t)i * R_DIM * D_DIM;

  // ---- Phase A: WMMA GEMM -------------------------------------------------
  const int wrow = wave * 16 + lane_lo;
  const __bf16* arow = cap + (size_t)imin(wrow, W_DIM - 1) * D_DIM;

  f32x8 acc[3] = {};
  for (int k0 = 0; k0 < D_DIM; k0 += 32) {
    bf16x16 a = load_a_frag(arow, k0, lane_hi);
#pragma unroll
    for (int rt = 0; rt < 3; ++rt) {
      const int nrow = rt * 16 + lane_lo;
      const __bf16* brow = img + (size_t)imin(nrow, R_DIM - 1) * D_DIM;
      bf16x16 b = load_b_frag(brow, k0, lane_hi);
      acc[rt] = __builtin_amdgcn_wmma_f32_16x16x32_bf16(false, a, false, b,
                                                        (short)0, acc[rt], false, false);
    }
  }

#pragma unroll
  for (int rt = 0; rt < 3; ++rt) {
#pragma unroll
    for (int j = 0; j < 8; ++j) {
      const int m = wave * 16 + j + 8 * lane_hi;   // local w row (0..63)
      const int n = rt * 16 + lane_lo;             // region r   (0..47)
      S[m * S_PITCH + n] = acc[rt][j];
    }
  }
  __syncthreads();

  // ---- Phase B: per-row epilogue -----------------------------------------
  if (threadIdx.x < W_DIM) {
    const int w  = threadIdx.x;
    const int cl = cap_lens[c];
    const int il = img_lens[i];
    const size_t oidx = ((size_t)i * N_CAP + c) * W_DIM + w;
    const float NEG_INF = -__builtin_inff();

    if (w >= cl) {
      out[oidx] = MASK_V;
    } else {
      float* Srow = &S[w * S_PITCH];

      // top-KNN values of sims = S_raw * (1/T), masked to -inf for r >= il
      float t5[KNN];
#pragma unroll
      for (int j = 0; j < KNN; ++j) t5[j] = NEG_INF;
      for (int r = 0; r < R_DIM; ++r) {
        float sims = (r < il) ? Srow[r] * INV_T : NEG_INF;
        if (sims > t5[KNN - 1]) {
          float v = sims;
#pragma unroll
          for (int j = 0; j < KNN; ++j) {
            if (v > t5[j]) { float tmp = t5[j]; t5[j] = v; v = tmp; }
          }
        }
      }
      const float thresh = t5[KNN - 1];
      float m = t5[0];
      if (!(m > NEG_INF)) m = 0.0f;    // where(isfinite(max), max, 0)

      // masked softmax numerators; num = sum p_r * S_raw[r]
      float ssum = 0.0f, numw = 0.0f;
      for (int r = 0; r < R_DIM; ++r) {
        const float raw  = Srow[r];
        const float sims = (r < il) ? raw * INV_T : NEG_INF;
        const float e    = (sims >= thresh) ? __expf(sims - m) : 0.0f;
        ssum += e;
        numw += e * raw;
        Srow[r] = e;                   // row owned exclusively by this thread
      }
      const float denom = (ssum == 0.0f) ? 1.0f : ssum;
      const float num   = numw / denom;

      // sq = p^T G p  (p is ~KNN-sparse, skip zero rows)
      float sqw = 0.0f;
      for (int r = 0; r < R_DIM; ++r) {
        const float er = Srow[r];
        if (er != 0.0f) {
          const float* grow = &Gs[r * G_PITCH];
          float accg = 0.0f;
          for (int r2 = 0; r2 < R_DIM; ++r2) accg += Srow[r2] * grow[r2];
          sqw += er * accg;
        }
      }
      const float sq   = sqw / (denom * denom);
      const float norm = (sq == 0.0f) ? 0.0f : sqrtf(sq);
      out[oidx] = num / (norm + EPS_V);
    }
  }
}

// ---------------------------------------------------------------------------
// Host launcher.  Workspace layout (all 256B aligned):
//   [0, 3276800)            caps as bf16 (64*50*512)
//   [3276800, 5636096)      imgs as bf16 (64*36*512)
//   [5636096, 5967872)      G    as f32  (64*36*36)
// ---------------------------------------------------------------------------
extern "C" void kernel_launch(void* const* d_in, const int* in_sizes, int n_in,
                              void* d_out, int out_size, void* d_ws, size_t ws_size,
                              hipStream_t stream) {
  const float* imgs     = (const float*)d_in[0];
  const float* caps     = (const float*)d_in[1];
  const int*   img_lens = (const int*)d_in[2];
  const int*   cap_lens = (const int*)d_in[3];
  float* out = (float*)d_out;

  char* wsp = (char*)d_ws;
  const size_t CAPS_BYTES = (size_t)N_CAP * W_DIM * D_DIM * 2;   // 3,276,800
  const size_t IMGS_BYTES = (size_t)N_IMG * R_DIM * D_DIM * 2;   // 2,359,296
  unsigned short* caps_bf = (unsigned short*)wsp;
  unsigned short* imgs_bf = (unsigned short*)(wsp + CAPS_BYTES);
  float*          G       = (float*)(wsp + CAPS_BYTES + IMGS_BYTES);

  const int total = N_CAP * W_DIM * D_DIM;  // max of the two element counts
  cvt_kernel<<<(total + 255) / 256, 256, 0, stream>>>(caps, imgs, caps_bf, imgs_bf);
  gram_kernel<<<N_IMG, 128, 0, stream>>>(imgs_bf, G);
  attn_kernel<<<dim3(N_CAP, N_IMG), 128, 0, stream>>>(caps_bf, imgs_bf, G,
                                                      img_lens, cap_lens, out);
}
